// isGNNNorm_65549790871885
// MI455X (gfx1250) — matrix operand order repaired and checked by reading, not compiled, not run on hardware
//
#include <hip/hip_runtime.h>
#include <hip/hip_bf16.h>

typedef __attribute__((ext_vector_type(16))) __bf16 v16bf;
typedef __attribute__((ext_vector_type(8)))  float  v8f;

#define CH 256          // feature channels per node
#define HEADS 8
#define HCH 32
#define NEG_SLOPE 0.2f
#define LN_EPS 1e-5f

// ---------------------------------------------------------------------------
// Weight pre-pass: f32 [K=256][N=256] row-major  ->  bf16 WMMA B-fragments.
// B (32x16 bf16 per WMMA): lanes 0-15 hold K{0..7,16..23} (2 per VGPR), lanes 16-31
// hold K{8..15,24..31}; column n -> lane (n&15). Fragment index:
//   (((ntile*8 + kstep)*32 + lane)*16 + elem)
// ---------------------------------------------------------------------------
__global__ __launch_bounds__(256) void swz_w_kernel(const float* __restrict__ W,
                                                    __bf16* __restrict__ out)
{
    int i = blockIdx.x * 256 + threadIdx.x;       // 0..65535
    int k = i >> 8, n = i & 255;
    int ntile = n >> 4, c = n & 15;
    int s = k >> 5, kk = k & 31;
    int lane = c + 16 * ((kk >> 3) & 1);
    int e = (kk & 7) + 8 * (kk >> 4);
    out[(((size_t)ntile * 8 + s) * 32 + lane) * 16 + e] = (__bf16)W[i];
}

// ---------------------------------------------------------------------------
// Y[M,256] = X[M,256] @ W[256,256]   (bf16 WMMA, f32 accumulate)
// grid.x = M/16 row tiles, grid.y = 2 column halves; 256 thr = 8 waves,
// each wave owns one 16x16 output tile.
// ---------------------------------------------------------------------------
__global__ __launch_bounds__(256) void gemm_wmma_kernel(const float* __restrict__ X,
                                                        const __bf16* __restrict__ Wsw,
                                                        float* __restrict__ Y)
{
    // A-fragment-ordered staging of 16 rows x 256 K in bf16 (8 KB LDS)
    __shared__ __bf16 sA[8 * 32 * 16];

    const int row0 = blockIdx.x * 16;
    const int tid  = threadIdx.x;

    // A (16x32 bf16 per WMMA): lanes 0-15 row M hold K{0..7,16..23} (2/VGPR),
    // lanes 16-31 hold K{8..15,24..31}.
    for (int i = tid; i < 4096; i += 256) {
        int r = i >> 8;            // 0..15
        int k = i & 255;           // 0..255
        float v = X[(size_t)(row0 + r) * CH + k];
        int s  = k >> 5;
        int kk = k & 31;
        int hv = (kk >> 3) & 1;                        // which half-lane group
        int kp = kk - 8 * hv;                          // {0..7} U {16..23}
        int vg = (kp < 8) ? (kp >> 1) : (4 + ((kp - 16) >> 1));
        int e  = vg * 2 + (kp & 1);
        int lane = r + 16 * hv;
        sA[(s * 32 + lane) * 16 + e] = (__bf16)v;
    }
    __syncthreads();

    const int wave = tid >> 5;
    const int lane = tid & 31;
    const int n0   = blockIdx.y * 128 + wave * 16;
    const int ntile = n0 >> 4;

    const __bf16* Bp = Wsw + ((size_t)ntile * 8) * 32 * 16;

    v8f acc = {};
    #pragma unroll
    for (int s = 0; s < 8; ++s) {
        v16bf a = *(const v16bf*)&sA[(s * 32 + lane) * 16];
        v16bf b = *(const v16bf*)&Bp[((size_t)s * 32 + lane) * 16];
        if (s < 7)
            __builtin_prefetch(&Bp[((size_t)(s + 1) * 32 + lane) * 16], 0, 3);
        acc = __builtin_amdgcn_wmma_f32_16x16x32_bf16(
                  false, a, false, b, (short)0, acc, false, false);
    }

    // C/D layout: VGPR g holds (M = g + 8*(lane>>4), N = lane&15)
    const int hv  = lane >> 4;
    const int col = lane & 15;
    union { v8f v; float f[8]; } u; u.v = acc;
    #pragma unroll
    for (int g = 0; g < 8; ++g)
        Y[(size_t)(row0 + g + 8 * hv) * CH + n0 + col] = u.f[g];
}

// ---------------------------------------------------------------------------
__global__ void fill_zero_kernel(float* __restrict__ p, long long n)
{
    long long i = (long long)blockIdx.x * blockDim.x + threadIdx.x;
    if (i < n) p[i] = 0.0f;
}

// order-preserving float->uint key for atomic max
__device__ __forceinline__ unsigned fkey_enc(float f) {
    unsigned u = __float_as_uint(f);
    return (u & 0x80000000u) ? ~u : (u | 0x80000000u);
}
__device__ __forceinline__ float fkey_dec(unsigned k) {
    return (k & 0x80000000u) ? __uint_as_float(k & 0x7FFFFFFFu)
                             : __uint_as_float(~k);
}

// ---------------------------------------------------------------------------
// score_eh = sum_c leakyrelu(xl[src,h,c] + xr[dst,h,c]) * att[h,c];
// segment max over dst via atomic on monotone uint key.
// ---------------------------------------------------------------------------
__global__ void edge_score_kernel(const int* __restrict__ srcA,
                                  const int* __restrict__ dstA,
                                  int E, int NN,
                                  const float* __restrict__ xl,
                                  const float* __restrict__ xr,
                                  const float* __restrict__ att,
                                  float* __restrict__ scores,
                                  unsigned* __restrict__ mkey)
{
    long long idx = (long long)blockIdx.x * blockDim.x + threadIdx.x;
    long long tot = (long long)(E + NN) * HEADS;
    if (idx >= tot) return;
    int e = (int)(idx >> 3), h = (int)(idx & 7);
    int s, d;
    if (e < E) { s = srcA[e]; d = dstA[e]; } else { s = d = e - E; }
    const float* pl = xl + (size_t)s * CH + h * HCH;
    const float* pr = xr + (size_t)d * CH + h * HCH;
    const float* pa = att + h * HCH;
    float sc = 0.f;
    #pragma unroll
    for (int c = 0; c < HCH; ++c) {
        float v = pl[c] + pr[c];
        v = (v > 0.f) ? v : NEG_SLOPE * v;
        sc += v * pa[c];
    }
    scores[idx] = sc;
    atomicMax(&mkey[(size_t)d * HEADS + h], fkey_enc(sc));
}

__global__ void edge_expsum_kernel(const int* __restrict__ dstA,
                                   int E, int NN,
                                   const unsigned* __restrict__ mkey,
                                   float* __restrict__ scores,
                                   float* __restrict__ denom)
{
    long long idx = (long long)blockIdx.x * blockDim.x + threadIdx.x;
    long long tot = (long long)(E + NN) * HEADS;
    if (idx >= tot) return;
    int e = (int)(idx >> 3), h = (int)(idx & 7);
    int d = (e < E) ? dstA[e] : (e - E);
    float m = fkey_dec(mkey[(size_t)d * HEADS + h]);
    float ex = __expf(scores[idx] - m);
    scores[idx] = ex;
    atomicAdd(&denom[(size_t)d * HEADS + h], ex);
}

__global__ void edge_aggr_kernel(const int* __restrict__ srcA,
                                 const int* __restrict__ dstA,
                                 int E, int NN,
                                 const float* __restrict__ scores,
                                 const float* __restrict__ denom,
                                 const float* __restrict__ xl,
                                 float* __restrict__ out)
{
    long long idx = (long long)blockIdx.x * blockDim.x + threadIdx.x;
    long long tot = (long long)(E + NN) * HEADS;
    if (idx >= tot) return;
    int e = (int)(idx >> 3), h = (int)(idx & 7);
    int s, d;
    if (e < E) { s = srcA[e]; d = dstA[e]; } else { s = d = e - E; }
    float alpha = scores[idx] / (denom[(size_t)d * HEADS + h] + 1e-16f);
    const float* pl = xl + (size_t)s * CH + h * HCH;
    float* po = out + (size_t)d * CH + h * HCH;
    #pragma unroll
    for (int c = 0; c < HCH; ++c)
        atomicAdd(&po[c], alpha * pl[c]);
}

// ---------------------------------------------------------------------------
// x = relu(LayerNorm(acc + bias) * g + b), in place; one block per node.
// ---------------------------------------------------------------------------
__global__ __launch_bounds__(256) void ln_relu_kernel(float* __restrict__ x,
                                                      const float* __restrict__ bias,
                                                      const float* __restrict__ g,
                                                      const float* __restrict__ b)
{
    __shared__ float red[CH];
    const int node = blockIdx.x;
    const int c = threadIdx.x;
    float v = x[(size_t)node * CH + c] + bias[c];
    red[c] = v;
    __syncthreads();
    for (int off = 128; off > 0; off >>= 1) {
        if (c < off) red[c] += red[c + off];
        __syncthreads();
    }
    float mu = red[0] * (1.0f / CH);
    __syncthreads();
    float dv = v - mu;
    red[c] = dv * dv;
    __syncthreads();
    for (int off = 128; off > 0; off >>= 1) {
        if (c < off) red[c] += red[c + off];
        __syncthreads();
    }
    float var = red[0] * (1.0f / CH);
    float r = dv * rsqrtf(var + LN_EPS) * g[c] + b[c];
    x[(size_t)node * CH + c] = (r > 0.f) ? r : 0.f;
}

// ---------------------------------------------------------------------------
// Ragged segment means, 3 modalities concatenated on channel dim.
// grid = (nseg, 3), 256 threads = one per channel.
// ---------------------------------------------------------------------------
__global__ __launch_bounds__(256) void seg_mean_kernel(const float* __restrict__ x,
                                                       const int* __restrict__ istns,
                                                       int nodesPerMod,
                                                       float* __restrict__ out)
{
    const int s = blockIdx.x;
    const int mod = blockIdx.y;
    const int c = threadIdx.x;
    const int lo = istns[s], hi = istns[s + 1];
    const size_t base = (size_t)mod * nodesPerMod;
    float acc = 0.f;
    for (int n = lo; n < hi; ++n)
        acc += x[(base + n) * CH + c];
    out[(size_t)s * (3 * CH) + mod * CH + c] = acc / (float)(hi - lo);
}

// ---------------------------------------------------------------------------
extern "C" void kernel_launch(void* const* d_in, const int* in_sizes, int n_in,
                              void* d_out, int out_size, void* d_ws, size_t ws_size,
                              hipStream_t stream)
{
    const float* x_in   = (const float*)d_in[0];
    const int*   edge   = (const int*)d_in[1];
    const int*   istns  = (const int*)d_in[6];

    const int NN   = in_sizes[0] / CH;     // 60000
    const int E    = in_sizes[1] / 2;      // 600000
    const int ETOT = E + NN;               // with self loops
    const int NSEG = in_sizes[6] - 1;      // 100
    const int M3   = NN / 3;               // 20000 nodes per modality

    const int* srcA = edge;
    const int* dstA = edge + E;

    // params flattened in setup_inputs insertion order:
    // per layer l: Wl, Wr, att, bias, g, b  (starting at d_in[8])
    const float *Wl[4], *Wr[4], *att[4], *bias[4], *gam[4], *bet[4];
    for (int l = 0; l < 4; ++l) {
        int p = 8 + l * 6;
        Wl[l]   = (const float*)d_in[p + 0];
        Wr[l]   = (const float*)d_in[p + 1];
        att[l]  = (const float*)d_in[p + 2];
        bias[l] = (const float*)d_in[p + 3];
        gam[l]  = (const float*)d_in[p + 4];
        bet[l]  = (const float*)d_in[p + 5];
    }

    // workspace layout (floats), all sub-regions 32B aligned
    const size_t NC = (size_t)NN * CH;
    float* ws    = (float*)d_ws;
    float* bufA  = ws;                 // ping
    float* bufB  = bufA + NC;          // pong
    float* xl    = bufB + NC;
    float* xr    = xl + NC;
    float* scores = xr + NC;                           // ETOT*8
    unsigned* mkey = (unsigned*)(scores + (size_t)ETOT * HEADS);   // NN*8
    float* denom   = (float*)(mkey + (size_t)NN * HEADS);          // NN*8
    __bf16* wswz   = (__bf16*)(denom + (size_t)NN * HEADS);        // 8 * 65536

    // ---- pre-swizzle all 8 projection weights to bf16 WMMA fragments ----
    for (int l = 0; l < 4; ++l) {
        swz_w_kernel<<<256, 256, 0, stream>>>(Wl[l], wswz + (size_t)(2 * l) * 65536);
        swz_w_kernel<<<256, 256, 0, stream>>>(Wr[l], wswz + (size_t)(2 * l + 1) * 65536);
    }

    const long long edgeThreads = (long long)ETOT * HEADS;
    const int edgeBlocks = (int)((edgeThreads + 255) / 256);
    const int zBlocksNC  = (int)((NC + 255) / 256);
    const int zBlocksNH  = (int)(((size_t)NN * HEADS + 255) / 256);
    dim3 gemmGrid(NN / 16, 2);

    for (int l = 0; l < 4; ++l) {
        const float* xin = (l == 0) ? x_in : ((l & 1) ? bufA : bufB);
        float* acc = (l & 1) ? bufB : bufA;    // l0->A, l1->B, l2->A, l3->B

        // projections
        gemm_wmma_kernel<<<gemmGrid, 256, 0, stream>>>(xin, wswz + (size_t)(2 * l) * 65536, xl);
        gemm_wmma_kernel<<<gemmGrid, 256, 0, stream>>>(xin, wswz + (size_t)(2 * l + 1) * 65536, xr);

        // clear accumulators
        fill_zero_kernel<<<zBlocksNC, 256, 0, stream>>>(acc, (long long)NC);
        fill_zero_kernel<<<zBlocksNH, 256, 0, stream>>>((float*)mkey, (long long)NN * HEADS);
        fill_zero_kernel<<<zBlocksNH, 256, 0, stream>>>(denom, (long long)NN * HEADS);

        // attention
        edge_score_kernel<<<edgeBlocks, 256, 0, stream>>>(srcA, dstA, E, NN, xl, xr,
                                                          att[l], scores, mkey);
        edge_expsum_kernel<<<edgeBlocks, 256, 0, stream>>>(dstA, E, NN, mkey, scores, denom);
        edge_aggr_kernel<<<edgeBlocks, 256, 0, stream>>>(srcA, dstA, E, NN, scores, denom,
                                                         xl, acc);

        // bias + layernorm + relu (in place)
        ln_relu_kernel<<<NN, 256, 0, stream>>>(acc, bias[l], gam[l], bet[l]);
    }

    // final features live in bufB (layer 4 output)
    dim3 segGrid(NSEG, 3);
    seg_mean_kernel<<<segGrid, 256, 0, stream>>>(bufB, istns, M3, (float*)d_out);
}
